// DIN_60859686584660
// MI455X (gfx1250) — matrix operand order repaired and checked by reading, not compiled, not run on hardware
//
#include <hip/hip_runtime.h>
#include <hip/hip_bf16.h>

typedef _Float16 v16h __attribute__((ext_vector_type(16)));
typedef float    v8f  __attribute__((ext_vector_type(8)));

union FragH { v16h v; float4 q[2]; };

// workspace layout (bytes, all 256-aligned)
#define WS_W1A   0          // 80*96  f16  = 15360  (din_w1 seq-part, transposed [n][k], k padded 80->96)
#define WS_MW1T  15360      // 128*352 f16 = 90112  (mlp_w1 transposed, k padded 336->352)
#define WS_MW2T  105472     // 128*128 f16 = 32768
#define WS_MW3T  138240     // 32*128  f16 = 8192
#define WS_MIN   146432     // 2048*352 f16 = 1441792 (padded mlp input rows)

__device__ __forceinline__ v8f wmma_acc(const FragH& a, const FragH& b, v8f c) {
  return __builtin_amdgcn_wmma_f32_16x16x32_f16(false, a.v, false, b.v, (short)0, c, false, false);
}

// ---------------------------------------------------------------------------
// Weight prep: f32 -> f16, transposed + K-padded for contiguous B-fragments
// ---------------------------------------------------------------------------
__global__ __launch_bounds__(256) void prep_kernel(
    const float* __restrict__ din_w1, const float* __restrict__ mlp_w1,
    const float* __restrict__ mlp_w2, const float* __restrict__ mlp_w3,
    _Float16* __restrict__ w1a, _Float16* __restrict__ w1t,
    _Float16* __restrict__ w2t, _Float16* __restrict__ w3t)
{
  int i = blockIdx.x * blockDim.x + threadIdx.x;
  if (i < 80 * 96) {                       // din W1a: [k<80 of 160, n<80] -> [n][k pad 96]
    int n = i / 96, k = i % 96;
    w1a[i] = (_Float16)(k < 80 ? din_w1[k * 80 + n] : 0.f);
    return;
  }
  i -= 80 * 96;
  if (i < 128 * 352) {                     // mlp w1: [336,128] -> [n][k pad 352]
    int n = i / 352, k = i % 352;
    w1t[i] = (_Float16)(k < 336 ? mlp_w1[k * 128 + n] : 0.f);
    return;
  }
  i -= 128 * 352;
  if (i < 128 * 128) {                     // mlp w2: [128,128] -> [n][k]
    int n = i / 128, k = i % 128;
    w2t[i] = (_Float16)mlp_w2[k * 128 + n];
    return;
  }
  i -= 128 * 128;
  if (i < 32 * 128) {                      // mlp w3: [128,32] -> [n][k]
    int n = i / 128, k = i % 128;
    w3t[i] = (_Float16)mlp_w3[k * 32 + n];
    return;
  }
}

// ---------------------------------------------------------------------------
// DIN attention: one workgroup (4 wave32) per batch row
// ---------------------------------------------------------------------------
__global__ __launch_bounds__(128) void din_kernel(
    const int* __restrict__ req_wday, const int* __restrict__ req_hour, const int* __restrict__ req_min,
    const int* __restrict__ uid, const int* __restrict__ did, const int* __restrict__ gender,
    const int* __restrict__ age, const int* __restrict__ province,
    const int* __restrict__ vid, const int* __restrict__ aid, const int* __restrict__ c2,
    const int* __restrict__ c1, const int* __restrict__ up,
    const int* __restrict__ uts_wday, const int* __restrict__ uts_hour, const int* __restrict__ uts_min,
    const int* __restrict__ seq_arr, const int* __restrict__ seq_mask,
    const float* __restrict__ uid_tab, const float* __restrict__ did_tab,
    const float* __restrict__ gender_tab, const float* __restrict__ age_tab,
    const float* __restrict__ province_tab,
    const float* __restrict__ vid_tab, const float* __restrict__ aid_tab,
    const float* __restrict__ c2_tab, const float* __restrict__ c1_tab,
    const float* __restrict__ up_tab,
    const float* __restrict__ wday_tab, const float* __restrict__ hour_tab,
    const float* __restrict__ min_tab,
    const float* __restrict__ din_w1, const float* __restrict__ din_b1,
    const float* __restrict__ din_w2, const float* __restrict__ din_b2,
    const _Float16* __restrict__ gW1a, _Float16* __restrict__ gMin)
{
  __shared__ __align__(16) _Float16 sSeq[208 * 96];  // seq emb, M pad 200->208, K pad 80->96
  __shared__ __align__(16) _Float16 sW[80 * 96];     // W1a transposed [n][k]
  __shared__ float sTgt[80];
  __shared__ float sTproj[80];
  __shared__ float sLogit[208];
  __shared__ float sScore[208];
  __shared__ float sW2[80];
  __shared__ float sRed[128];

  const int tid  = threadIdx.x;
  const int b    = blockIdx.x;
  const int lane = tid & 31;
  const int wave = tid >> 5;
  const int mrow = lane & 15;
  const int hi   = lane >> 4;   // half-wave select (ISA fragment layout)

  // ---- stage: zero padded seq tile, copy W1a, target embedding, w2 ----
  for (int i = tid; i < (208 * 96) / 2; i += 128) ((unsigned int*)sSeq)[i] = 0u;
  for (int i = tid; i < (80 * 96) / 2; i += 128)
    ((unsigned int*)sW)[i] = ((const unsigned int*)gW1a)[i];
  if (tid < 80) sW2[tid] = din_w2[tid];
  if (tid < 80) {
    int f = tid >> 4, e = tid & 15, id;
    const float* tab;
    switch (f) {
      case 0:  tab = vid_tab; id = vid[b]; break;
      case 1:  tab = aid_tab; id = aid[b]; break;
      case 2:  tab = c2_tab;  id = c2[b];  break;
      case 3:  tab = c1_tab;  id = c1[b];  break;
      default: tab = up_tab;  id = up[b];  break;
    }
    sTgt[tid] = tab[(size_t)id * 16 + e];
  }
  __syncthreads();

  // ---- gather 200x5x16 history embeddings -> f16 LDS (float4 gathers) ----
  {
    const int base = b * 1000;                     // B,S,5
    for (int i = tid; i < 4000; i += 128) {        // (p,f,quad) tasks
      int p = i / 20, r = i % 20, f = r >> 2, q = r & 3;
      const float* tab;
      switch (f) {
        case 0:  tab = vid_tab; break;
        case 1:  tab = aid_tab; break;
        case 2:  tab = c2_tab;  break;
        case 3:  tab = c1_tab;  break;
        default: tab = up_tab;  break;
      }
      int id = seq_arr[base + p * 5 + f];
      float4 v = *(const float4*)(tab + (size_t)id * 16 + q * 4);
      union { float2 f2; _Float16 h[4]; } o;
      o.h[0] = (_Float16)v.x; o.h[1] = (_Float16)v.y;
      o.h[2] = (_Float16)v.z; o.h[3] = (_Float16)v.w;
      *(float2*)&sSeq[p * 96 + f * 16 + q * 4] = o.f2;
    }
  }
  // ---- target projection through W1b rows (computed once per batch row) ----
  if (tid < 80) {
    float acc = din_b1[tid];
    for (int k = 0; k < 80; ++k) acc += sTgt[k] * din_w1[(80 + k) * 80 + tid];
    sTproj[tid] = acc;
  }
  __syncthreads();

  // ---- layer1 GEMM (208x80x96) + fused relu -> logits, via v_wmma f16 ----
  const float b2 = din_b2[0];
  for (int mt = wave; mt < 13; mt += 4) {          // each wave owns disjoint M-tiles
    const int p = mt * 16 + mrow;
    FragH a0, a1, a2;
    // 16-bit A layout: lane holds K = kb2+{0..7} and kb2+16+{0..7}, kb2 = kbase + hi*8
    a0.q[0] = *(const float4*)&sSeq[p * 96 + 0  + hi * 8];
    a0.q[1] = *(const float4*)&sSeq[p * 96 + 16 + hi * 8];
    a1.q[0] = *(const float4*)&sSeq[p * 96 + 32 + hi * 8];
    a1.q[1] = *(const float4*)&sSeq[p * 96 + 48 + hi * 8];
    a2.q[0] = *(const float4*)&sSeq[p * 96 + 64 + hi * 8];
    a2.q[1] = *(const float4*)&sSeq[p * 96 + 80 + hi * 8];
    float plog[8];
#pragma unroll
    for (int j = 0; j < 8; ++j) plog[j] = 0.f;
#pragma unroll
    for (int nt = 0; nt < 5; ++nt) {
      const int n = nt * 16 + mrow;
      v8f acc = {0.f, 0.f, 0.f, 0.f, 0.f, 0.f, 0.f, 0.f};
      FragH bf;
      // 16-bit B layout: lane holds K = kb2+{0..7} and kb2+16+{0..7}, kb2 = kbase + hi*8
      bf.q[0] = *(const float4*)&sW[n * 96 + 0  + hi * 8];
      bf.q[1] = *(const float4*)&sW[n * 96 + 16 + hi * 8];
      acc = wmma_acc(a0, bf, acc);
      bf.q[0] = *(const float4*)&sW[n * 96 + 32 + hi * 8];
      bf.q[1] = *(const float4*)&sW[n * 96 + 48 + hi * 8];
      acc = wmma_acc(a1, bf, acc);
      bf.q[0] = *(const float4*)&sW[n * 96 + 64 + hi * 8];
      bf.q[1] = *(const float4*)&sW[n * 96 + 80 + hi * 8];
      acc = wmma_acc(a2, bf, acc);
      const float tp = sTproj[n];
      const float w2 = sW2[n];
#pragma unroll
      for (int j = 0; j < 8; ++j) {
        float h = fmaxf(acc[j] + tp, 0.f);         // relu(h) fused
        plog[j] += h * w2;                         // partial logit over this lane's n
      }
    }
    // reduce logits across the 16 lanes of each half-wave (n dimension)
#pragma unroll
    for (int j = 0; j < 8; ++j) {
#pragma unroll
      for (int off = 8; off > 0; off >>= 1)
        plog[j] += __shfl_xor(plog[j], off, 16);
    }
    if (mrow == 0) {
#pragma unroll
      for (int j = 0; j < 8; ++j)
        sLogit[mt * 16 + hi * 8 + j] = plog[j] + b2;
    }
  }
  __syncthreads();

  // ---- masked softmax over S=200 ----
  const int* msk = seq_mask + (size_t)b * 200;
  float lm = -3.0e38f;
  for (int p = tid; p < 200; p += 128) {
    float lg = (msk[p] != 0) ? sLogit[p] : -1073741823.0f;  // -2^30+1 pad
    sLogit[p] = lg;
    lm = fmaxf(lm, lg);
  }
  sRed[tid] = lm; __syncthreads();
  for (int s = 64; s > 0; s >>= 1) {
    if (tid < s) sRed[tid] = fmaxf(sRed[tid], sRed[tid + s]);
    __syncthreads();
  }
  const float gmax = sRed[0]; __syncthreads();
  float ls = 0.f;
  for (int p = tid; p < 200; p += 128) {
    float e = __expf(sLogit[p] - gmax);
    sScore[p] = e; ls += e;
  }
  sRed[tid] = ls; __syncthreads();
  for (int s = 64; s > 0; s >>= 1) {
    if (tid < s) sRed[tid] += sRed[tid + s];
    __syncthreads();
  }
  const float inv = 1.f / sRed[0];
  for (int p = tid; p < 200; p += 128) sScore[p] *= inv;
  if (tid < 8) sScore[200 + tid] = 0.f;
  __syncthreads();

  // ---- pooled interest + assemble padded f16 mlp input row [352] ----
  _Float16* orow = gMin + (size_t)b * 352;
  for (int c = tid; c < 352; c += 128) {
    float v;
    if (c < 128) {                                  // user/context embeddings
      int f = c >> 4, e = c & 15, id;
      const float* tab;
      switch (f) {
        case 0:  tab = wday_tab;     id = req_wday[b]; break;
        case 1:  tab = hour_tab;     id = req_hour[b]; break;
        case 2:  tab = min_tab;      id = req_min[b];  break;
        case 3:  tab = uid_tab;      id = uid[b];      break;
        case 4:  tab = did_tab;      id = did[b];      break;
        case 5:  tab = gender_tab;   id = gender[b];   break;
        case 6:  tab = age_tab;      id = age[b];      break;
        default: tab = province_tab; id = province[b]; break;
      }
      v = tab[(size_t)id * 16 + e];
    } else if (c < 208) {                           // din interest = scores @ seq_emb
      int k = c - 128;
      float acc = 0.f;
      for (int p = 0; p < 200; ++p) acc += sScore[p] * (float)sSeq[p * 96 + k];
      v = acc;
    } else if (c < 288) {                           // target embeddings
      v = sTgt[c - 208];
    } else if (c < 336) {                           // upload-ts embeddings
      int u = c - 288, f = u >> 4, e = u & 15, id;
      const float* tab;
      switch (f) {
        case 0:  tab = wday_tab; id = uts_wday[b]; break;
        case 1:  tab = hour_tab; id = uts_hour[b]; break;
        default: tab = min_tab;  id = uts_min[b];  break;
      }
      v = tab[(size_t)id * 16 + e];
    } else v = 0.f;                                 // K padding 336->352
    orow[c] = (_Float16)v;
  }
}

// ---------------------------------------------------------------------------
// Final MLP: 16 batch rows per workgroup, all hidden layers via WMMA
// ---------------------------------------------------------------------------
__global__ __launch_bounds__(128) void mlp_kernel(
    const _Float16* __restrict__ gMin,
    const _Float16* __restrict__ gW1t, const float* __restrict__ b1,
    const _Float16* __restrict__ gW2t, const float* __restrict__ b2,
    const _Float16* __restrict__ gW3t, const float* __restrict__ b3,
    const float* __restrict__ w4, const float* __restrict__ b4,
    float* __restrict__ out)
{
  __shared__ __align__(16) _Float16 sIn[16 * 352];
  __shared__ __align__(16) _Float16 sH1[16 * 128];
  __shared__ __align__(16) _Float16 sH2[16 * 128];
  __shared__ __align__(16) _Float16 sH3[16 * 32];

  const int tid  = threadIdx.x;
  const int lane = tid & 31;
  const int wave = tid >> 5;
  const int mrow = lane & 15;
  const int hi   = lane >> 4;
  const int b0   = blockIdx.x * 16;

  const float4* src = (const float4*)(gMin + (size_t)b0 * 352);
  for (int i = tid; i < (16 * 352) / 8; i += 128) ((float4*)sIn)[i] = src[i];
  __syncthreads();

  // layer1: 352 -> 128
  for (int nt = wave; nt < 8; nt += 4) {
    const int n = nt * 16 + mrow;
    v8f acc = {0.f, 0.f, 0.f, 0.f, 0.f, 0.f, 0.f, 0.f};
    for (int kk = 0; kk < 11; ++kk) {
      FragH a, bf;
      const int kb2 = kk * 32 + hi * 8;
      a.q[0] = *(const float4*)&sIn[mrow * 352 + kb2];
      a.q[1] = *(const float4*)&sIn[mrow * 352 + kb2 + 16];
      const int ko = kk * 32 + hi * 8;
      bf.q[0] = *(const float4*)&gW1t[(size_t)n * 352 + ko];
      bf.q[1] = *(const float4*)&gW1t[(size_t)n * 352 + ko + 16];
      acc = wmma_acc(a, bf, acc);
    }
    const float bias = b1[n];
#pragma unroll
    for (int j = 0; j < 8; ++j)
      sH1[(j + hi * 8) * 128 + n] = (_Float16)fmaxf(acc[j] + bias, 0.f);
  }
  __syncthreads();

  // layer2: 128 -> 128
  for (int nt = wave; nt < 8; nt += 4) {
    const int n = nt * 16 + mrow;
    v8f acc = {0.f, 0.f, 0.f, 0.f, 0.f, 0.f, 0.f, 0.f};
    for (int kk = 0; kk < 4; ++kk) {
      FragH a, bf;
      const int kb2 = kk * 32 + hi * 8;
      a.q[0] = *(const float4*)&sH1[mrow * 128 + kb2];
      a.q[1] = *(const float4*)&sH1[mrow * 128 + kb2 + 16];
      const int ko = kk * 32 + hi * 8;
      bf.q[0] = *(const float4*)&gW2t[(size_t)n * 128 + ko];
      bf.q[1] = *(const float4*)&gW2t[(size_t)n * 128 + ko + 16];
      acc = wmma_acc(a, bf, acc);
    }
    const float bias = b2[n];
#pragma unroll
    for (int j = 0; j < 8; ++j)
      sH2[(j + hi * 8) * 128 + n] = (_Float16)fmaxf(acc[j] + bias, 0.f);
  }
  __syncthreads();

  // layer3: 128 -> 32 (waves 0,1 only — wave-uniform)
  for (int nt = wave; nt < 2; nt += 4) {
    const int n = nt * 16 + mrow;
    v8f acc = {0.f, 0.f, 0.f, 0.f, 0.f, 0.f, 0.f, 0.f};
    for (int kk = 0; kk < 4; ++kk) {
      FragH a, bf;
      const int kb2 = kk * 32 + hi * 8;
      a.q[0] = *(const float4*)&sH2[mrow * 128 + kb2];
      a.q[1] = *(const float4*)&sH2[mrow * 128 + kb2 + 16];
      const int ko = kk * 32 + hi * 8;
      bf.q[0] = *(const float4*)&gW3t[(size_t)n * 128 + ko];
      bf.q[1] = *(const float4*)&gW3t[(size_t)n * 128 + ko + 16];
      acc = wmma_acc(a, bf, acc);
    }
    const float bias = b3[n];
#pragma unroll
    for (int j = 0; j < 8; ++j)
      sH3[(j + hi * 8) * 32 + n] = (_Float16)fmaxf(acc[j] + bias, 0.f);
  }
  __syncthreads();

  // layer4: 32 -> 1
  if (tid < 16) {
    float acc = b4[0];
    for (int k = 0; k < 32; ++k) acc += (float)sH3[tid * 32 + k] * w4[k];
    out[b0 + tid] = acc;
  }
}

// ---------------------------------------------------------------------------
extern "C" void kernel_launch(void* const* d_in, const int* in_sizes, int n_in,
                              void* d_out, int out_size, void* d_ws, size_t ws_size,
                              hipStream_t stream) {
  (void)in_sizes; (void)n_in; (void)out_size; (void)ws_size;
  char* ws = (char*)d_ws;
  _Float16* w1a  = (_Float16*)(ws + WS_W1A);
  _Float16* w1t  = (_Float16*)(ws + WS_MW1T);
  _Float16* w2t  = (_Float16*)(ws + WS_MW2T);
  _Float16* w3t  = (_Float16*)(ws + WS_MW3T);
  _Float16* gmin = (_Float16*)(ws + WS_MIN);

  // total prep elements: 80*96 + 128*352 + 128*128 + 32*128 = 73216 = 286*256
  prep_kernel<<<286, 256, 0, stream>>>(
      (const float*)d_in[31], (const float*)d_in[35],
      (const float*)d_in[37], (const float*)d_in[39],
      w1a, w1t, w2t, w3t);

  din_kernel<<<2048, 128, 0, stream>>>(
      (const int*)d_in[0],  (const int*)d_in[1],  (const int*)d_in[2],
      (const int*)d_in[3],  (const int*)d_in[4],  (const int*)d_in[5],
      (const int*)d_in[6],  (const int*)d_in[7],
      (const int*)d_in[8],  (const int*)d_in[9],  (const int*)d_in[10],
      (const int*)d_in[11], (const int*)d_in[12],
      (const int*)d_in[13], (const int*)d_in[14], (const int*)d_in[15],
      (const int*)d_in[16], (const int*)d_in[17],
      (const float*)d_in[18], (const float*)d_in[19], (const float*)d_in[20],
      (const float*)d_in[21], (const float*)d_in[22],
      (const float*)d_in[23], (const float*)d_in[24], (const float*)d_in[25],
      (const float*)d_in[26], (const float*)d_in[27],
      (const float*)d_in[28], (const float*)d_in[29], (const float*)d_in[30],
      (const float*)d_in[31], (const float*)d_in[32],
      (const float*)d_in[33], (const float*)d_in[34],
      w1a, gmin);

  mlp_kernel<<<128, 128, 0, stream>>>(
      gmin,
      w1t, (const float*)d_in[36],
      w2t, (const float*)d_in[38],
      w3t, (const float*)d_in[40],
      (const float*)d_in[41], (const float*)d_in[42],
      (float*)d_out);
}
